// Attention_74698071212133
// MI455X (gfx1250) — matrix operand order repaired and checked by reading, not compiled, not run on hardware
//
#include <hip/hip_runtime.h>
#include <hip/hip_bf16.h>
#include <math.h>

typedef __bf16 bf16_t;
typedef __attribute__((ext_vector_type(16))) __bf16 v16bf;
typedef __attribute__((ext_vector_type(8)))  __bf16 v8bf;
typedef __attribute__((ext_vector_type(8)))  float  v8f;

#define B_   512
#define T_   64
#define D_   512
#define H_   512
#define C_   38
#define S_   26
#define K_   512           // every GEMM in this model has K == 512
#define G4H  (4*H_)
#define DCW  (D_ + C_)     // 550 = w_ih row stride
#define NPAD 48            // C padded to 3 WMMA n-tiles

// ---------------- WMMA fragment loaders (wave32 layouts per CDNA5 ISA 7.12.2) ---

// A: 16x32 bf16 tile, row-major with leading dim = K_. lane&15 = M row;
// elements 0..7 -> K = (lane>>4)*8 + 0..7 ; elements 8..15 -> K = 16 + (lane>>4)*8 + 0..7
static __device__ __forceinline__ v16bf load_a_frag(const bf16_t* __restrict__ A, int lane) {
  const int row = lane & 15, half = lane >> 4;
  const bf16_t* p = A + (size_t)row * K_ + half * 8;
  v8bf lo = *(const v8bf*)p;
  v8bf hi = *(const v8bf*)(p + 16);
  return __builtin_shufflevector(lo, hi, 0,1,2,3,4,5,6,7,8,9,10,11,12,13,14,15);
}

// B: 32x16 bf16 tile (K x N). We compute A @ W^T with W stored [N,K] row-major,
// so column n of B is row n of W (contiguous). lane&15 = N col;
// elements 0..7 -> K = (lane>>4)*8 + 0..7 ; elements 8..15 -> K = 16 + (lane>>4)*8 + 0..7
static __device__ __forceinline__ v16bf load_b_frag(const bf16_t* __restrict__ Wrow, int lane) {
  const int half = lane >> 4;
  const bf16_t* p = Wrow + half * 8;
  v8bf lo = *(const v8bf*)p;
  v8bf hi = *(const v8bf*)(p + 16);
  return __builtin_shufflevector(lo, hi, 0,1,2,3,4,5,6,7,8,9,10,11,12,13,14,15);
}

// ---------------- Generic WMMA GEMM: C[M,N] (+)= A[M,512] @ W[N,512]^T --------
// grid.x = M/16, each wave owns a 16 x (NT*16) tile of C.
template<int NT, bool ACC, bool BIAS>
__global__ void wmma_gemm_k512(const bf16_t* __restrict__ A,
                               const bf16_t* __restrict__ W,
                               float* __restrict__ Cmat,
                               const float* __restrict__ bias,
                               int Ntot, int ldc, int ncols) {
  const int lane = threadIdx.x & 31;
  const int wave = threadIdx.x >> 5;
  const int wavesPerBlk = blockDim.x >> 5;
  const int m0 = blockIdx.x * 16;
  const int n0 = (blockIdx.y * wavesPerBlk + wave) * (NT * 16);
  if (n0 >= Ntot) return;
  const int row = lane & 15, half = lane >> 4;

  v8f acc[NT];
#pragma unroll
  for (int t = 0; t < NT; ++t) {
    if (ACC) {
#pragma unroll
      for (int i = 0; i < 8; ++i)
        acc[t][i] = Cmat[(size_t)(m0 + i + 8*half) * ldc + n0 + t*16 + row];
    } else {
#pragma unroll
      for (int i = 0; i < 8; ++i) acc[t][i] = 0.0f;
    }
  }

  const bf16_t* Ap = A + (size_t)m0 * K_;
#pragma unroll 4
  for (int k0 = 0; k0 < K_; k0 += 32) {
    v16bf a = load_a_frag(Ap + k0, lane);
#pragma unroll
    for (int t = 0; t < NT; ++t) {
      const bf16_t* Wrow = W + (size_t)(n0 + t*16 + row) * K_ + k0;
      v16bf b = load_b_frag(Wrow, lane);
      acc[t] = __builtin_amdgcn_wmma_f32_16x16x32_bf16(
          false, a, false, b, (short)0, acc[t], false, false);
    }
  }

#pragma unroll
  for (int t = 0; t < NT; ++t) {
    const int n = n0 + t*16 + row;
    if (n < ncols) {
      float bv = BIAS ? bias[n] : 0.0f;
#pragma unroll
      for (int i = 0; i < 8; ++i)
        Cmat[(size_t)(m0 + i + 8*half) * ldc + n] = acc[t][i] + bv;
    }
  }
}

// ---------------- Fused attention step (score + softmax + context) ------------
__global__ __launch_bounds__(256)
void attention_step(const float* __restrict__ Hproj, const float* __restrict__ hp,
                    const float* __restrict__ b_h2h, const float* __restrict__ wscore,
                    const float* __restrict__ enc, bf16_t* __restrict__ ctx_b) {
  __shared__ float hps[H_];
  __shared__ float wss[H_];
  __shared__ float es[T_];
  __shared__ float als[T_];
  const int b = blockIdx.x;
  const int tid = threadIdx.x;
  for (int h = tid; h < H_; h += 256) {
    hps[h] = hp[(size_t)b * H_ + h] + b_h2h[h];
    wss[h] = wscore[h];
  }
  __syncthreads();

  const int wave = tid >> 5, lane = tid & 31;
  const float* Hb = Hproj + (size_t)b * T_ * H_;
  // each of the 8 waves scores 8 encoder positions
  for (int j = 0; j < 8; ++j) {
    const int t = wave * 8 + j;
    const float* Ht = Hb + (size_t)t * H_;
    float s = 0.f;
#pragma unroll 4
    for (int h = lane; h < H_; h += 32)
      s += tanhf(Ht[h] + hps[h]) * wss[h];
    for (int off = 16; off > 0; off >>= 1) s += __shfl_down(s, off, 32);
    if (lane == 0) es[t] = s;
  }
  __syncthreads();

  // softmax over T=64 in wave 0
  if (tid < 32) {
    float e0 = es[tid], e1 = es[tid + 32];
    float m = fmaxf(e0, e1);
    for (int off = 16; off > 0; off >>= 1) m = fmaxf(m, __shfl_down(m, off, 32));
    m = __shfl(m, 0, 32);
    float p0 = expf(e0 - m), p1 = expf(e1 - m);
    float sum = p0 + p1;
    for (int off = 16; off > 0; off >>= 1) sum += __shfl_down(sum, off, 32);
    sum = __shfl(sum, 0, 32);
    const float inv = 1.f / sum;
    als[tid] = p0 * inv;
    als[tid + 32] = p1 * inv;
  }
  __syncthreads();

  // context = alpha @ enc[b]
  const float* Eb = enc + (size_t)b * T_ * D_;
  for (int d = tid; d < D_; d += 256) {
    float s = 0.f;
#pragma unroll 8
    for (int t = 0; t < T_; ++t) s += als[t] * Eb[(size_t)t * D_ + d];
    ctx_b[(size_t)b * D_ + d] = (bf16_t)s;
  }
}

// ---------------- LSTM pointwise (+ one-hot column gather + biases) -----------
static __device__ __forceinline__ float sigf(float x) { return 1.f / (1.f + expf(-x)); }

__global__ __launch_bounds__(256)
void lstm_pointwise(const float* __restrict__ gates, const float* __restrict__ w_ih,
                    const int* __restrict__ text, int s,
                    const float* __restrict__ b_ih, const float* __restrict__ b_hh,
                    float* __restrict__ c, bf16_t* __restrict__ h_b,
                    bf16_t* __restrict__ hidden_b) {
  const int idx = blockIdx.x * 256 + threadIdx.x;      // B_*H_ threads
  const int b = idx >> 9, hh = idx & (H_ - 1);
  const int cls = text[b * S_ + s];
  const float* gb = gates + (size_t)b * G4H;
  const float ig = gb[hh]          + b_ih[hh]          + b_hh[hh]          + w_ih[(size_t)hh * DCW + D_ + cls];
  const float fg = gb[H_ + hh]     + b_ih[H_ + hh]     + b_hh[H_ + hh]     + w_ih[(size_t)(H_ + hh) * DCW + D_ + cls];
  const float gg = gb[2*H_ + hh]   + b_ih[2*H_ + hh]   + b_hh[2*H_ + hh]   + w_ih[(size_t)(2*H_ + hh) * DCW + D_ + cls];
  const float og = gb[3*H_ + hh]   + b_ih[3*H_ + hh]   + b_hh[3*H_ + hh]   + w_ih[(size_t)(3*H_ + hh) * DCW + D_ + cls];
  const float cn = sigf(fg) * c[idx] + sigf(ig) * tanhf(gg);
  const float hn = sigf(og) * tanhf(cn);
  c[idx] = cn;
  h_b[idx] = (bf16_t)hn;
  hidden_b[((size_t)b * S_ + s) * H_ + hh] = (bf16_t)hn;
}

// ---------------- conversion / init kernels -----------------------------------
__global__ void cvt_f32_bf16(const float* __restrict__ in, bf16_t* __restrict__ out, int n) {
  for (int i = blockIdx.x * blockDim.x + threadIdx.x; i < n; i += gridDim.x * blockDim.x)
    out[i] = (bf16_t)in[i];
}

__global__ void pack_w_ihc(const float* __restrict__ w_ih, bf16_t* __restrict__ out) {
  const int i = blockIdx.x * 256 + threadIdx.x;        // G4H*D_
  if (i >= G4H * D_) return;
  const int r = i >> 9, k = i & (D_ - 1);
  out[i] = (bf16_t)w_ih[(size_t)r * DCW + k];
}

__global__ void pack_w_gen(const float* __restrict__ w, bf16_t* __restrict__ out) {
  const int i = blockIdx.x * 256 + threadIdx.x;        // NPAD*H_
  if (i >= NPAD * H_) return;
  const int r = i >> 9, k = i & (H_ - 1);
  out[i] = (r < C_) ? (bf16_t)w[(size_t)r * H_ + k] : (bf16_t)0.f;
}

__global__ void init_state(float* __restrict__ c, bf16_t* __restrict__ h) {
  const int i = blockIdx.x * 256 + threadIdx.x;
  if (i < B_ * H_) { c[i] = 0.f; h[i] = (bf16_t)0.f; }
}

// ---------------- host orchestration ------------------------------------------
extern "C" void kernel_launch(void* const* d_in, const int* in_sizes, int n_in,
                              void* d_out, int out_size, void* d_ws, size_t ws_size,
                              hipStream_t stream) {
  const float* enc    = (const float*)d_in[0];
  const int*   text   = (const int*)  d_in[1];
  const float* w_i2h  = (const float*)d_in[2];
  const float* w_h2h  = (const float*)d_in[3];
  const float* b_h2h  = (const float*)d_in[4];
  const float* w_scr  = (const float*)d_in[5];
  const float* w_ih   = (const float*)d_in[6];
  const float* w_hh   = (const float*)d_in[7];
  const float* b_ih   = (const float*)d_in[8];
  const float* b_hh   = (const float*)d_in[9];
  const float* w_gen  = (const float*)d_in[10];
  const float* b_gen  = (const float*)d_in[11];
  float* probs = (float*)d_out;

  char* wsp = (char*)d_ws;
  auto alloc = [&](size_t bytes) { char* p = wsp; wsp += (bytes + 255) & ~(size_t)255; return p; };

  bf16_t* enc_b    = (bf16_t*)alloc((size_t)B_*T_*D_*sizeof(bf16_t));
  float*  Hproj    = (float*) alloc((size_t)B_*T_*H_*sizeof(float));
  bf16_t* w_i2h_b  = (bf16_t*)alloc((size_t)H_*D_*sizeof(bf16_t));
  bf16_t* w_h2h_b  = (bf16_t*)alloc((size_t)H_*H_*sizeof(bf16_t));
  bf16_t* w_ihc_b  = (bf16_t*)alloc((size_t)G4H*D_*sizeof(bf16_t));
  bf16_t* w_hh_b   = (bf16_t*)alloc((size_t)G4H*H_*sizeof(bf16_t));
  bf16_t* w_gen_b  = (bf16_t*)alloc((size_t)NPAD*H_*sizeof(bf16_t));
  bf16_t* h_b      = (bf16_t*)alloc((size_t)B_*H_*sizeof(bf16_t));
  bf16_t* ctx_b    = (bf16_t*)alloc((size_t)B_*D_*sizeof(bf16_t));
  float*  c_st     = (float*) alloc((size_t)B_*H_*sizeof(float));
  float*  hp       = (float*) alloc((size_t)B_*H_*sizeof(float));
  float*  gates    = (float*) alloc((size_t)B_*G4H*sizeof(float));
  bf16_t* hid_b    = (bf16_t*)alloc((size_t)B_*S_*H_*sizeof(bf16_t));

  // one-time conversions
  cvt_f32_bf16<<<4096, 256, 0, stream>>>(enc,   enc_b,   B_*T_*D_);
  cvt_f32_bf16<<<1024, 256, 0, stream>>>(w_i2h, w_i2h_b, H_*D_);
  cvt_f32_bf16<<<1024, 256, 0, stream>>>(w_h2h, w_h2h_b, H_*H_);
  cvt_f32_bf16<<<4096, 256, 0, stream>>>(w_hh,  w_hh_b,  G4H*H_);
  pack_w_ihc<<<(G4H*D_ + 255)/256, 256, 0, stream>>>(w_ih, w_ihc_b);
  pack_w_gen<<<(NPAD*H_ + 255)/256, 256, 0, stream>>>(w_gen, w_gen_b);
  init_state<<<(B_*H_ + 255)/256, 256, 0, stream>>>(c_st, h_b);

  // H_proj = enc @ w_i2h^T   [32768 x 512] = [32768 x 512] x [512 x 512]^T
  wmma_gemm_k512<4, false, false><<<dim3((B_*T_)/16, 2), 128, 0, stream>>>(
      enc_b, w_i2h_b, Hproj, nullptr, H_, H_, H_);

  for (int s = 0; s < S_; ++s) {
    // hp = h @ w_h2h^T   (bias folded in attention kernel)
    wmma_gemm_k512<4, false, false><<<dim3(B_/16, 2), 128, 0, stream>>>(
        h_b, w_h2h_b, hp, nullptr, H_, H_, H_);
    // attention: scores + softmax + context
    attention_step<<<B_, 256, 0, stream>>>(Hproj, hp, b_h2h, w_scr, enc, ctx_b);
    // gates = ctx @ w_ih[:, :D]^T ...
    wmma_gemm_k512<4, false, false><<<dim3(B_/16, 8), 128, 0, stream>>>(
        ctx_b, w_ihc_b, gates, nullptr, G4H, G4H, G4H);
    // ... += h @ w_hh^T
    wmma_gemm_k512<4, true, false><<<dim3(B_/16, 8), 128, 0, stream>>>(
        h_b, w_hh_b, gates, nullptr, G4H, G4H, G4H);
    // LSTM cell update (one-hot column gather + biases folded here)
    lstm_pointwise<<<(B_*H_)/256, 256, 0, stream>>>(
        gates, w_ih, text, s, b_ih, b_hh, c_st, h_b, hid_b);
  }

  // probs = hiddens @ w_gen^T + b_gen  [13312 x 38], N padded to 48, masked store
  wmma_gemm_k512<3, false, true><<<dim3((B_*S_)/16, 1), 32, 0, stream>>>(
      hid_b, w_gen_b, probs, b_gen, NPAD, C_, C_);
}
